// QKVAttentionLegacy_14516989460771
// MI455X (gfx1250) — matrix operand order repaired and checked by reading, not compiled, not run on hardware
//
#include <hip/hip_runtime.h>
#include <math.h>

typedef _Float16 v16h __attribute__((ext_vector_type(16)));
typedef _Float16 h8   __attribute__((ext_vector_type(8)));
typedef float    v8f  __attribute__((ext_vector_type(8)));

#define N_HEADS 8
#define BS 4
#define T 2048
#define CH 64
#define BH (BS * N_HEADS)          // 32 batched heads
#define HEAD_STRIDE (3 * CH * T)   // 192*2048 floats per head in qkv

#define S_TILE 32
#define WAVES 8
#define M_TILES 2                  // two 16-row query tiles per wave
#define M_PER_WAVE (16 * M_TILES)  // 32
#define T_PER_BLOCK (WAVES * M_PER_WAVE)   // 256
#define KT_STRIDE 72    // halves per s-row of K^T tile (16B-aligned b128 frags, conflict-spread)
#define V_STRIDE  40    // halves per c-row of V tile
#define P_STRIDE  32    // halves per t-row of the per-wave P slab

__device__ __forceinline__ v8f wmma_f16(v16h a, v16h b, v8f c) {
  // D = A(16x32 f16) x B(32x16 f16) + C(16x16 f32)
  return __builtin_amdgcn_wmma_f32_16x16x32_f16(false, a, false, b, (short)0, c, false, false);
}

__global__ __launch_bounds__(256, 1)
void flash_attn_f16wmma_kernel(const float* __restrict__ qkv,
                               float* __restrict__ out) {
  __shared__ _Float16 KT[S_TILE * KT_STRIDE];                        // [s][c]  32 x 72
  __shared__ _Float16 Vsh[CH * V_STRIDE];                            // [c][s]  64 x 40
  __shared__ _Float16 Psh[WAVES * M_TILES * 16 * P_STRIDE];          // per-wave 32 x 32

  const int tid  = threadIdx.x;
  const int wave = tid >> 5;
  const int lane = tid & 31;
  const int lh   = lane >> 4;      // lane half (0/1)
  const int ln   = lane & 15;

  const int bid = blockIdx.x;
  const int bh  = bid >> 3;                       // 0..31 : head index
  const int tb  = bid & 7;                        // t-block
  const int t0  = tb * T_PER_BLOCK + wave * M_PER_WAVE;

  // Entire softmax scaling folded into Q:  S' = (q*k)/8 * log2(e);  P = exp2(S') = exp(q*k/8).
  const float qscale = 0.125f * 1.4426950408889634f;   // (1/8) * log2(e)

  const float* qb = qkv + (size_t)bh * HEAD_STRIDE;          // rows 0..63   : Q
  const float* kb = qb + (size_t)CH * T;                     // rows 64..127 : K
  const float* vb = qb + (size_t)(2 * CH) * T;               // rows 128..191: V
  float* ob = out + (size_t)bh * CH * T;

  // ---- Q A-fragments: 32x64 tile -> 2 Mtiles x 2 chalves of 16x32 f16 frags ----
  // A frags, contiguous k order chosen here (B frags match): lane m = l%16; K = half*16 + 2v{+1}
  v16h aq[M_TILES][2];
  #pragma unroll
  for (int mt = 0; mt < M_TILES; ++mt) {
    const int t = t0 + mt * 16 + ln;
    #pragma unroll
    for (int v = 0; v < 8; ++v) {
      const int grp = v >> 2, j = v & 3;
      const int c = lh * 16 + 2 * v;
      aq[mt][0][2*v]   = (_Float16)(qb[(size_t)c        * T + t] * qscale);
      aq[mt][0][2*v+1] = (_Float16)(qb[(size_t)(c + 1)  * T + t] * qscale);
      aq[mt][1][2*v]   = (_Float16)(qb[(size_t)(c + 32) * T + t] * qscale);
      aq[mt][1][2*v+1] = (_Float16)(qb[(size_t)(c + 33) * T + t] * qscale);
    }
  }

  // all-ones B fragment: row-sum of P via WMMA (every column of L = rowsum)
  v16h ones;
  #pragma unroll
  for (int i = 0; i < 16; ++i) ones[i] = (_Float16)1.0f;

  v8f oacc[M_TILES][4];     // O tiles 16x64 (4 n-tiles each), f32
  v8f lacc[M_TILES];        // row-sum accumulators
  #pragma unroll
  for (int mt = 0; mt < M_TILES; ++mt) {
    lacc[mt] = v8f{};
    #pragma unroll
    for (int ct = 0; ct < 4; ++ct) oacc[mt][ct] = v8f{};
  }

  _Float16* Pw = Psh + wave * (M_TILES * 16 * P_STRIDE);   // wave-private 2KB

  for (int s0 = 0; s0 < T; s0 += S_TILE) {
    // ---- cooperative staging: K (transposed, unscaled) and V into LDS as f16 ----
    #pragma unroll
    for (int i = 0; i < 4; ++i) {
      const int c = wave * 8 + 2 * i + lh;
      const float2 xk = *(const float2*)&kb[(size_t)c * T + s0 + 2 * ln];
      KT[(2 * ln)     * KT_STRIDE + c] = (_Float16)xk.x;
      KT[(2 * ln + 1) * KT_STRIDE + c] = (_Float16)xk.y;
      const float2 xv = *(const float2*)&vb[(size_t)c * T + s0 + 2 * ln];
      Vsh[c * V_STRIDE + 2 * ln]     = (_Float16)xv.x;
      Vsh[c * V_STRIDE + 2 * ln + 1] = (_Float16)xv.y;
    }
    // prefetch next s-tile (address wrapped, never OOB, no divergent guard)
    {
      const int s0n = (s0 + S_TILE) & (T - 1);
      const int c = wave * 8 + lh;
      __builtin_prefetch(&kb[(size_t)c * T + s0n + 2 * ln], 0, 1);
      __builtin_prefetch(&vb[(size_t)c * T + s0n + 2 * ln], 0, 1);
    }
    __syncthreads();

    // ---- K B-fragments (shared by both M tiles) ----
    // B frags, same contiguous k order as the Q A-frags: lane n = l%16; K = lhalf*16 + 2v{+1} (+ chalf*32)
    union { v16h v; h8 h[2]; } bk[2][2];   // [ntile][chalf]
    #pragma unroll
    for (int nt = 0; nt < 2; ++nt) {
      const int ns = nt * 16 + ln;
      bk[nt][0].h[0] = *(const h8*)&KT[ns * KT_STRIDE + lh * 16 + 0];
      bk[nt][0].h[1] = *(const h8*)&KT[ns * KT_STRIDE + lh * 16 + 8];
      bk[nt][1].h[0] = *(const h8*)&KT[ns * KT_STRIDE + 32 + lh * 16 + 0];
      bk[nt][1].h[1] = *(const h8*)&KT[ns * KT_STRIDE + 32 + lh * 16 + 8];
    }

    // ---- S' = (Q*qscale)^T K ; P = exp2(S') (no max subtraction needed) ----
    #pragma unroll
    for (int mt = 0; mt < M_TILES; ++mt) {
      #pragma unroll
      for (int nt = 0; nt < 2; ++nt) {
        v8f c = {};
        c = wmma_f16(aq[mt][0], bk[nt][0].v, c);
        c = wmma_f16(aq[mt][1], bk[nt][1].v, c);
        // C layout: VGPR r holds row r + 8*lh, column = ln
        #pragma unroll
        for (int r = 0; r < 8; ++r) {
          const int trow = mt * 16 + r + 8 * lh;
          Pw[trow * P_STRIDE + nt * 16 + ln] = (_Float16)__builtin_amdgcn_exp2f(c[r]);
        }
      }
    }
    asm volatile("s_wait_dscnt 0" ::: "memory");

    // ---- V B-fragments (shared by both M tiles) ----
    union { v16h v; h8 h[2]; } bv[4];
    #pragma unroll
    for (int ct = 0; ct < 4; ++ct) {
      const int nc = ct * 16 + ln;
      bv[ct].h[0] = *(const h8*)&Vsh[nc * V_STRIDE + lh * 16 + 0];
      bv[ct].h[1] = *(const h8*)&Vsh[nc * V_STRIDE + lh * 16 + 8];
    }

    // ---- reload P as A-fragments; O += P V^T ; L += P * ones ----
    #pragma unroll
    for (int mt = 0; mt < M_TILES; ++mt) {
      union { v16h v; h8 h[2]; } ap;
      const _Float16* Pm = Pw + mt * 16 * P_STRIDE;
      ap.h[0] = *(const h8*)&Pm[ln * P_STRIDE + lh * 16];
      ap.h[1] = *(const h8*)&Pm[ln * P_STRIDE + lh * 16 + 8];
      lacc[mt] = wmma_f16(ap.v, ones, lacc[mt]);
      #pragma unroll
      for (int ct = 0; ct < 4; ++ct) {
        oacc[mt][ct] = wmma_f16(ap.v, bv[ct].v, oacc[mt][ct]);
      }
    }
    __syncthreads();
  }

  // ---- epilogue: normalize by reciprocal row-sum, transpose via LDS, coalesced stores ----
  float* Ow = (float*)Pw;   // reuse 1KB of the wave-private slab (16x16 f32)
  #pragma unroll
  for (int mt = 0; mt < M_TILES; ++mt) {
    float rinv[8];
    #pragma unroll
    for (int r = 0; r < 8; ++r) rinv[r] = __builtin_amdgcn_rcpf(lacc[mt][r]);
    #pragma unroll
    for (int ct = 0; ct < 4; ++ct) {
      #pragma unroll
      for (int r = 0; r < 8; ++r) {
        const int trow = r + 8 * lh;
        Ow[ln * 16 + trow] = oacc[mt][ct][r] * rinv[r];
      }
      asm volatile("s_wait_dscnt 0" ::: "memory");
      const float4 x0 = *(const float4*)&Ow[ln * 16 + lh * 8];
      const float4 x1 = *(const float4*)&Ow[ln * 16 + lh * 8 + 4];
      float* dst = ob + (size_t)(ct * 16 + ln) * T + t0 + mt * 16 + lh * 8;
      *(float4*)(dst)     = x0;
      *(float4*)(dst + 4) = x1;
      asm volatile("s_wait_dscnt 0" ::: "memory");
    }
  }
}

extern "C" void kernel_launch(void* const* d_in, const int* in_sizes, int n_in,
                              void* d_out, int out_size, void* d_ws, size_t ws_size,
                              hipStream_t stream) {
  const float* qkv = (const float*)d_in[0];
  float* out = (float*)d_out;
  dim3 grid(BH * (T / T_PER_BLOCK));   // 32 heads * 8 t-blocks = 256 workgroups
  dim3 block(256);                     // 8 wave32 waves
  hipLaunchKernelGGL(flash_attn_f16wmma_kernel, grid, block, 0, stream, qkv, out);
}
